// GNNPOLY_7911329759796
// MI455X (gfx1250) — matrix operand order repaired and checked by reading, not compiled, not run on hardware
//
#include <hip/hip_runtime.h>
#include <hip/hip_bf16.h>

typedef __attribute__((ext_vector_type(16))) _Float16 v16h;
typedef __attribute__((ext_vector_type(8)))  float    v8f;

union AU { v16h v; uint4 q[2]; };
union HPack { _Float16 h[2]; unsigned u; };

__device__ __forceinline__ void atomicAddF(float* p, float v) {
    __hip_atomic_fetch_add(p, v, __ATOMIC_RELAXED, __HIP_MEMORY_SCOPE_AGENT);
}

#define LOG2E 1.4426950408889634f
#define LN2   0.6931471805599453f

// sigmoid(xf) * softplus(xs) with raw HW transcendentals:
// v_exp_f32 is base-2; log argument is in [1,2] so raw v_log_f32 needs no fixup.
__device__ __forceinline__ float gate_act(float xf, float xs) {
    float sig = __builtin_amdgcn_rcpf(
        1.0f + __builtin_amdgcn_exp2f(-xf * LOG2E));
    float u   = __builtin_amdgcn_exp2f(-fabsf(xs) * LOG2E);
    float sp  = fmaxf(xs, 0.0f) + LN2 * __builtin_amdgcn_logf(1.0f + u);
    return sig * sp;
}

// ---------------------------------------------------------------------------
// Prep: build padded/permuted B fragments for the 4 edge-weight matrices.
// z K-order: [0..9]=h_dst (W cols 0..9), [10..12]=e (W cols 20..22),
//            [13..15]=0, [16..25]=h_src (W cols 10..19), [26..31]=0.
// B (32x16 f16) layout: lane<16 -> col n=lane, half t holds K=t;
//                       lane>=16 -> col n=lane-16, half t holds K=16+t.
// Buffer layout: mat*512 + lane*16 + t.
// ---------------------------------------------------------------------------
__global__ void prep_bfrag_kernel(const float* __restrict__ f1,
                                  const float* __restrict__ s1,
                                  const float* __restrict__ f2,
                                  const float* __restrict__ s2,
                                  _Float16* __restrict__ bfrag) {
    int t = blockIdx.x * blockDim.x + threadIdx.x;
    if (t >= 4 * 32 * 16) return;
    int mat  = t >> 9;
    int lane = (t >> 4) & 31;
    int h    = t & 15;
    int n    = lane & 15;
    int K    = (lane < 16) ? h : (16 + h);
    const float* W = (mat == 0) ? f1 : (mat == 1) ? s1 : (mat == 2) ? f2 : s2;
    float v = 0.0f;
    if (n < 10) {
        int col = -1;
        if (K < 10)                 col = K;             // x_i block
        else if (K < 13)            col = 20 + (K - 10); // edge attr block
        else if (K >= 16 && K < 26) col = 10 + (K - 16); // x_j block
        if (col >= 0) v = W[n * 23 + col];
    }
    bfrag[t] = (_Float16)v;
}

// ---------------------------------------------------------------------------
// Node pre-layer: h = relu(x @ pre_W.T + pre_b), store f32 + f16 (padded to
// 16 features), and zero-init the aggregation buffer.
// ---------------------------------------------------------------------------
__global__ void pre_kernel(const float* __restrict__ x,
                           const float* __restrict__ W,
                           const float* __restrict__ b,
                           float* __restrict__ h32,
                           _Float16* __restrict__ h16,
                           float* __restrict__ agg, int N) {
    int n = blockIdx.x * blockDim.x + threadIdx.x;
    if (n >= N) return;
    float x0 = x[n * 3 + 0], x1 = x[n * 3 + 1], x2 = x[n * 3 + 2];
    #pragma unroll
    for (int o = 0; o < 16; ++o) {
        float v = 0.0f;
        if (o < 10) {
            v = fmaf(W[o * 3 + 0], x0,
                fmaf(W[o * 3 + 1], x1,
                fmaf(W[o * 3 + 2], x2, b[o])));
            v = fmaxf(v, 0.0f);
        }
        h32[(size_t)n * 16 + o] = v;
        h16[(size_t)n * 16 + o] = (_Float16)v;
        agg[(size_t)n * 16 + o] = 0.0f;
    }
}

// ---------------------------------------------------------------------------
// Edge layer: 16 edges per wave -> one 16x32 f16 A tile (K=23 padded to 32),
// two v_wmma_f32_16x16x32_f16 (gate + softplus weights, bias folded into C),
// fast activations, f32 atomic scatter into agg[dst].
// ---------------------------------------------------------------------------
__global__ void __launch_bounds__(256)
edge_kernel(const _Float16* __restrict__ h16,
            const long long* __restrict__ ei,   // [2, E]: row0 = src, row1 = dst
            const float* __restrict__ ea,       // [E, 3]
            const _Float16* __restrict__ bfragF,
            const _Float16* __restrict__ bfragS,
            const float* __restrict__ biasF,
            const float* __restrict__ biasS,
            float* __restrict__ agg, int E) {
    __shared__ __align__(16) _Float16 zbuf[8][16 * 32];  // per-wave 16 edges x 32 halves
    __shared__ __align__(16) unsigned dstbuf[8][16];

    const int lane  = threadIdx.x & 31;
    const int wave  = threadIdx.x >> 5;
    const int gwave = blockIdx.x * (blockDim.x >> 5) + wave;
    const int nwave = gridDim.x * (blockDim.x >> 5);
    const bool lo   = lane < 16;
    const int  m    = lane & 15;   // edge row within tile
    const int  n    = m;           // output feature column for this lane

    // B fragments: held in VGPRs across all tiles.
    AU Bf, Bs;
    Bf.v = *(const v16h*)(bfragF + lane * 16);
    Bs.v = *(const v16h*)(bfragS + lane * 16);

    // Biases folded into the WMMA C operand: C[m][n] = bias[n] (lane-uniform).
    const float bF = (n < 10) ? biasF[n] : 0.0f;
    const float bS = (n < 10) ? biasS[n] : 0.0f;
    v8f cF, cS;
    #pragma unroll
    for (int r = 0; r < 8; ++r) { cF[r] = bF; cS[r] = bS; }

    const int ntile = (E + 15) >> 4;
    for (int tile = gwave; tile < ntile; tile += nwave) {
        const int e  = tile * 16 + m;
        const int ec = (e < E) ? e : (E - 1);

        // lane m gathers h[dst] (+edge attr); lane m+16 gathers h[src]
        const long long nodeIdx = lo ? ei[(size_t)E + ec] : ei[ec];
        const _Float16* hp = h16 + ((size_t)(unsigned)nodeIdx << 4);
        uint4    hA = *(const uint4*)hp;            // halves 0..7
        unsigned hB = *(const unsigned*)(hp + 8);   // halves 8..9

        unsigned* zu = (unsigned*)(&zbuf[wave][m * 32]);
        if (lo) {
            ((uint4*)zu)[0] = hA;                   // K 0..7
            zu[4] = hB;                             // K 8..9
            HPack p5, p6;
            const float* ep = ea + (size_t)ec * 3;
            p5.h[0] = (_Float16)ep[0]; p5.h[1] = (_Float16)ep[1];
            p6.h[0] = (_Float16)ep[2]; p6.h[1] = (_Float16)0.0f;
            zu[5] = p5.u;                           // K 10..11
            zu[6] = p6.u;                           // K 12 (+pad)
            zu[7] = 0u;                             // K 14..15
            dstbuf[wave][m] = (unsigned)nodeIdx;
        } else {
            unsigned* zh = zu + 8;                  // halves 16..31
            ((uint4*)zh)[0] = hA;                   // K 16..23
            zh[4] = hB;                             // K 24..25
            zh[5] = 0u; zh[6] = 0u; zh[7] = 0u;     // K 26..31
        }

        // Cross-lane LDS dependence within the wave: wait for all DS stores.
        asm volatile("s_wait_dscnt 0x0" ::: "memory");

        // A fragment in contiguous K order, same as the B fragment:
        // lane<16: K0..15 (v0-7); lane>=16: K16..31.
        const char* zb = (const char*)(&zbuf[wave][0]) + m * 64 + (lo ? 0 : 32);
        AU A;
        A.q[0] = *(const uint4*)zb;
        A.q[1] = *(const uint4*)(zb + 16);

        // Destination indices for this half-wave's 8 rows: two b128 DS loads,
        // covered by the same s_wait_dscnt, kept in VGPRs for the unrolled loop.
        const uint4* db = (const uint4*)(&dstbuf[wave][lo ? 0 : 8]);
        const uint4 dA = db[0], dB = db[1];

        v8f df = __builtin_amdgcn_wmma_f32_16x16x32_f16(
            false, A.v, false, Bf.v, (short)0, cF, false, false);
        v8f dsv = __builtin_amdgcn_wmma_f32_16x16x32_f16(
            false, A.v, false, Bs.v, (short)0, cS, false, false);

        // D layout: vgpr r -> edge (lo ? r : 8+r), column n = lane&15.
        if (n < 10) {
            const unsigned didx[8] = { dA.x, dA.y, dA.z, dA.w,
                                       dB.x, dB.y, dB.z, dB.w };
            if (tile * 16 + 16 <= E) {
                // Fast path: whole tile in range -> no per-r exec masking.
                #pragma unroll
                for (int r = 0; r < 8; ++r) {
                    float val = gate_act(df[r], dsv[r]);
                    atomicAddF(agg + ((didx[r] << 4) | (unsigned)n), val);
                }
            } else {
                #pragma unroll
                for (int r = 0; r < 8; ++r) {
                    const int eg = tile * 16 + (lo ? r : (8 + r));
                    float val = gate_act(df[r], dsv[r]);
                    if (eg < E)
                        atomicAddF(agg + ((didx[r] << 4) | (unsigned)n), val);
                }
            }
        }
        // WAR guard before the next iteration rewrites zbuf.
        asm volatile("s_wait_dscnt 0x0" ::: "memory");
    }
}

// h = relu(h + agg); refresh f16 copy; reset agg for the next layer.
__global__ void update_kernel(float* __restrict__ h32,
                              _Float16* __restrict__ h16,
                              float* __restrict__ agg, int NF) {
    int i = blockIdx.x * blockDim.x + threadIdx.x;
    if (i >= NF) return;
    float v = fmaxf(h32[i] + agg[i], 0.0f);
    h32[i] = v;
    h16[i] = (_Float16)v;
    agg[i] = 0.0f;
}

__global__ void pool_init_kernel(float* __restrict__ gsum,
                                 float* __restrict__ cnt, int G) {
    int i = blockIdx.x * blockDim.x + threadIdx.x;
    if (i < G * 16) gsum[i] = 0.0f;
    if (i < G) cnt[i] = 0.0f;
}

__global__ void pool_kernel(const float* __restrict__ h32,
                            const long long* __restrict__ batch,
                            float* __restrict__ gsum,
                            float* __restrict__ cnt, int N) {
    int nd = blockIdx.x * blockDim.x + threadIdx.x;
    if (nd >= N) return;
    unsigned b = (unsigned)batch[nd];
    #pragma unroll
    for (int f = 0; f < 10; ++f)
        atomicAddF(gsum + ((b << 4) | (unsigned)f), h32[(size_t)nd * 16 + f]);
    atomicAddF(cnt + b, 1.0f);
}

__global__ void mean_kernel(const float* __restrict__ gsum,
                            const float* __restrict__ cnt,
                            float* __restrict__ gmean, int G) {
    int t = blockIdx.x * blockDim.x + threadIdx.x;
    if (t >= G * 16) return;
    gmean[t] = gsum[t] * __builtin_amdgcn_rcpf(fmaxf(cnt[t >> 4], 1.0f));
}

__global__ void fc_kernel(const float* __restrict__ in,
                          const float* __restrict__ W,
                          const float* __restrict__ b,
                          float* __restrict__ out,
                          int G, int IN, int INSTRIDE, int OUT, int relu) {
    int t = blockIdx.x * blockDim.x + threadIdx.x;
    if (t >= G * OUT) return;
    int o = t % OUT, g = t / OUT;
    const float* ip = in + (size_t)g * INSTRIDE;
    const float* wp = W + (size_t)o * IN;
    float acc = b[o];
    for (int k = 0; k < IN; ++k) acc = fmaf(ip[k], wp[k], acc);
    if (relu) acc = fmaxf(acc, 0.0f);
    out[t] = acc;
}

extern "C" void kernel_launch(void* const* d_in, const int* in_sizes, int n_in,
                              void* d_out, int out_size, void* d_ws, size_t ws_size,
                              hipStream_t stream) {
    const float*     x     = (const float*)d_in[0];
    const long long* ei    = (const long long*)d_in[1];
    const float*     ea    = (const float*)d_in[2];
    const long long* batch = (const long long*)d_in[3];
    const float* pre_W = (const float*)d_in[4];  const float* pre_b = (const float*)d_in[5];
    const float* f1_W  = (const float*)d_in[6];  const float* f1_b  = (const float*)d_in[7];
    const float* s1_W  = (const float*)d_in[8];  const float* s1_b  = (const float*)d_in[9];
    const float* f2_W  = (const float*)d_in[10]; const float* f2_b  = (const float*)d_in[11];
    const float* s2_W  = (const float*)d_in[12]; const float* s2_b  = (const float*)d_in[13];
    const float* fc1_W = (const float*)d_in[14]; const float* fc1_b = (const float*)d_in[15];
    const float* fc2_W = (const float*)d_in[16]; const float* fc2_b = (const float*)d_in[17];
    const float* out_W = (const float*)d_in[18]; const float* out_b = (const float*)d_in[19];
    (void)n_in; (void)ws_size;

    const int N = in_sizes[0] / 3;
    const int E = in_sizes[2] / 3;
    const int G = out_size / 3;

    // Carve workspace (~33 MB total).
    char* p = (char*)d_ws;
    auto carve = [&](size_t bytes) {
        char* r = p;
        p += (bytes + 255) & ~(size_t)255;
        return (void*)r;
    };
    float*     h32   = (float*)carve((size_t)N * 16 * 4);
    _Float16*  h16   = (_Float16*)carve((size_t)N * 16 * 2);
    float*     agg   = (float*)carve((size_t)N * 16 * 4);
    _Float16*  bfrag = (_Float16*)carve(4 * 32 * 16 * 2);
    float*     gsum  = (float*)carve((size_t)G * 16 * 4);
    float*     cnt   = (float*)carve((size_t)G * 4);
    float*     gmean = (float*)carve((size_t)G * 16 * 4);
    float*     g1    = (float*)carve((size_t)G * 128 * 4);
    float*     g2    = (float*)carve((size_t)G * 128 * 4);

    prep_bfrag_kernel<<<(2048 + 255) / 256, 256, 0, stream>>>(f1_W, s1_W, f2_W, s2_W, bfrag);
    pre_kernel<<<(N + 255) / 256, 256, 0, stream>>>(x, pre_W, pre_b, h32, h16, agg, N);

    const int eblocks = 2048;
    // layer 1
    edge_kernel<<<eblocks, 256, 0, stream>>>(h16, ei, ea, bfrag + 0 * 512, bfrag + 1 * 512,
                                             f1_b, s1_b, agg, E);
    update_kernel<<<(N * 16 + 255) / 256, 256, 0, stream>>>(h32, h16, agg, N * 16);
    // layer 2
    edge_kernel<<<eblocks, 256, 0, stream>>>(h16, ei, ea, bfrag + 2 * 512, bfrag + 3 * 512,
                                             f2_b, s2_b, agg, E);
    update_kernel<<<(N * 16 + 255) / 256, 256, 0, stream>>>(h32, h16, agg, N * 16);

    // global mean pool
    pool_init_kernel<<<(G * 16 + 255) / 256, 256, 0, stream>>>(gsum, cnt, G);
    pool_kernel<<<(N + 255) / 256, 256, 0, stream>>>(h32, batch, gsum, cnt, N);
    mean_kernel<<<(G * 16 + 255) / 256, 256, 0, stream>>>(gsum, cnt, gmean, G);

    // MLP head (f32)
    fc_kernel<<<(G * 128 + 255) / 256, 256, 0, stream>>>(gmean, fc1_W, fc1_b, g1, G, 10, 16, 128, 1);
    fc_kernel<<<(G * 128 + 255) / 256, 256, 0, stream>>>(g1, fc2_W, fc2_b, g2, G, 128, 128, 128, 1);
    fc_kernel<<<(G * 3 + 255) / 256, 256, 0, stream>>>(g2, out_W, out_b, (float*)d_out, G, 128, 128, 3, 0);
}
